// MultiHeadAttention_34299608826069
// MI455X (gfx1250) — compile-verified, matrix operand order checked by reading, not run on hardware
//
#include <hip/hip_runtime.h>
#include <hip/hip_bf16.h>

typedef __bf16 bf16;
typedef __attribute__((ext_vector_type(16))) __bf16 v16bf;
typedef __attribute__((ext_vector_type(8)))  __bf16 v8bf;
typedef __attribute__((ext_vector_type(8)))  float  v8f;

#define D_MODEL 2048
#define SEQ     2048
#define NHEADS  16
#define DKDIM   128
#define BATCH   2
#define MROWS   (BATCH * SEQ)   /* 4096 */

// ---------- helpers ----------

__device__ __forceinline__ bf16 f2bf(float f) {
  union { float f; unsigned u; } v; v.f = f;
  unsigned r = v.u + 0x7FFFu + ((v.u >> 16) & 1u);   // round-to-nearest-even
  unsigned short h = (unsigned short)(r >> 16);
  return __builtin_bit_cast(bf16, h);
}

__device__ __forceinline__ v8f zero8() {
  v8f z;
#pragma unroll
  for (int i = 0; i < 8; ++i) z[i] = 0.0f;
  return z;
}

__device__ __forceinline__ v8f wmma_bf16(v16bf a, v16bf b, v8f c) {
  // D = A(16x32 bf16) * B(32x16 bf16) + C(16x16 f32)
  return __builtin_amdgcn_wmma_f32_16x16x32_bf16(
      false, a, false, b, (short)0, c, false, false);
}

// Load a 16x32 bf16 A-fragment (also used as B-fragment of the transpose).
// Memory is K-major with row stride `ldk` elements; `p0` points at [row0, k0].
// Per ISA 7.12.2: lane L holds row L&15; halves 0..7 are K = hi*8 + 0..7,
// halves 8..15 are K = 16 + hi*8 + 0..7, hi = L>>4.
__device__ __forceinline__ v16bf load_frag(const bf16* __restrict__ p0, int ldk) {
  const int lane = threadIdx.x & 31;
  const int r    = lane & 15;
  const int h8   = (lane >> 4) << 3;
  const bf16* p  = p0 + (size_t)r * ldk + h8;
  v8bf lo = *(const v8bf*)(p);
  v8bf hi = *(const v8bf*)(p + 16);
  v16bf f;
#pragma unroll
  for (int i = 0; i < 8; ++i) { f[i] = lo[i]; f[i + 8] = hi[i]; }
  return f;
}

__device__ __forceinline__ float rmax16(float v) {
#pragma unroll
  for (int m = 1; m < 16; m <<= 1) v = fmaxf(v, __shfl_xor(v, m, 16));
  return v;
}
__device__ __forceinline__ float rsum16(float v) {
#pragma unroll
  for (int m = 1; m < 16; m <<= 1) v += __shfl_xor(v, m, 16);
  return v;
}

// ---------- fp32 -> bf16 convert ----------

__global__ void f32_to_bf16_kernel(const float* __restrict__ in,
                                   bf16* __restrict__ out, int n) {
  int i = blockIdx.x * blockDim.x + threadIdx.x;
  int stride = gridDim.x * blockDim.x;
  for (; i < n; i += stride) out[i] = f2bf(in[i]);
}

// ---------- GEMM: C[M,N] = A[M,K] * W[N,K]^T + bias[N] ----------
// Wave tile 64x64 (4x4 WMMA frags): 16 WMMA per 8 fragment loads.
// Block = 8 waves arranged 2(M) x 4(N) -> block tile 128 x 256.
// mode 0: fp32 row-major [M,N] -> outF
// mode 1: bf16 [B,H,S,D]       -> outB  (Q, K)
// mode 2: bf16 [B,H,D,S]       -> outB  (V transposed)

__global__ __launch_bounds__(256)
void gemm_bf16_kernel(const bf16* __restrict__ A, const bf16* __restrict__ W,
                      const float* __restrict__ bias,
                      float* __restrict__ outF, bf16* __restrict__ outB,
                      int M, int N, int K, int mode) {
  const int lane = threadIdx.x & 31;
  const int wave = threadIdx.x >> 5;
  const int wm   = wave & 1;           // 0..1
  const int wn   = wave >> 1;          // 0..3
  const int m0   = blockIdx.y * 128 + wm * 64;
  const int n0   = blockIdx.x * 256 + wn * 64;
  const int r    = lane & 15;
  const int h8   = (lane >> 4) << 3;

  v8f c[4][4];
#pragma unroll
  for (int i = 0; i < 4; ++i)
#pragma unroll
    for (int j = 0; j < 4; ++j) c[i][j] = zero8();

  const bf16* Abase = A + (size_t)m0 * K;
  const bf16* Wbase = W + (size_t)n0 * K;

  for (int k0 = 0; k0 < K; k0 += 32) {
    v16bf a[4], b[4];
#pragma unroll
    for (int i = 0; i < 4; ++i) a[i] = load_frag(Abase + (size_t)(16 * i) * K + k0, K);
#pragma unroll
    for (int j = 0; j < 4; ++j) b[j] = load_frag(Wbase + (size_t)(16 * j) * K + k0, K);
#pragma unroll
    for (int i = 0; i < 4; ++i)
#pragma unroll
      for (int j = 0; j < 4; ++j)
        c[i][j] = wmma_bf16(a[i], b[j], c[i][j]);
  }

#pragma unroll
  for (int ci = 0; ci < 4; ++ci) {
#pragma unroll
    for (int cj = 0; cj < 4; ++cj) {
      const int col = n0 + cj * 16 + r;
      const float bv = bias[col];
#pragma unroll
      for (int i = 0; i < 8; ++i) {
        const float v = c[ci][cj][i] + bv;
        const int row = m0 + ci * 16 + i + h8;
        if (mode == 0) {
          outF[(size_t)row * N + col] = v;
        } else {
          const int bb = row >> 11;       // row / SEQ
          const int s  = row & (SEQ - 1);
          const int h  = col >> 7;        // col / DKDIM
          const int d  = col & (DKDIM - 1);
          if (mode == 1)
            outB[(((size_t)(bb * NHEADS + h)) * SEQ + s) * DKDIM + d] = f2bf(v);
          else
            outB[(((size_t)(bb * NHEADS + h)) * DKDIM + d) * SEQ + s] = f2bf(v);
        }
      }
    }
  }
}

// ---------- flash attention (causal) ----------
// Q,K: bf16 [B,H,S,D]; Vt: bf16 [B,H,D,S]; O: bf16 [B,S,H*D]
// One wave handles a 16-row Q tile; keys processed in chunks of 32.

__global__ __launch_bounds__(256)
void attn_kernel(const bf16* __restrict__ Q, const bf16* __restrict__ Kk,
                 const bf16* __restrict__ Vt, bf16* __restrict__ O) {
  __shared__ __align__(16) bf16 pbuf[8][16 * 32];   // per-wave 16x32 P tile

  const int wave = threadIdx.x >> 5;
  const int lane = threadIdx.x & 31;
  const int gw   = blockIdx.x * 8 + wave;
  const int qtiles = SEQ / 16;                 // 128
  const int bh   = gw / qtiles;
  const int q0   = (gw % qtiles) * 16;
  const int r    = lane & 15;
  const int h8   = (lane >> 4) << 3;

  // Preload full Q tile (16 x 128) as 4 fragments.
  const bf16* qbase = Q + ((size_t)bh * SEQ + q0) * DKDIM;
  v16bf qf[4];
#pragma unroll
  for (int j = 0; j < 4; ++j) qf[j] = load_frag(qbase + j * 32, DKDIM);

  v8f of[8];
#pragma unroll
  for (int j = 0; j < 8; ++j) of[j] = zero8();
  float m[8], l[8];
#pragma unroll
  for (int i = 0; i < 8; ++i) { m[i] = -1e30f; l[i] = 0.0f; }

  const float sc = 0.08838834764831845f;       // 1/sqrt(128)

  for (int t0 = 0; t0 < q0 + 16; t0 += 32) {
    // --- scores: two 16x16 tiles over keys [t0, t0+31] ---
    v8f s0 = zero8(), s1 = zero8();
#pragma unroll
    for (int j = 0; j < 4; ++j) {
      v16bf kf = load_frag(Kk + ((size_t)bh * SEQ + t0) * DKDIM + j * 32, DKDIM);
      s0 = wmma_bf16(qf[j], kf, s0);
    }
#pragma unroll
    for (int j = 0; j < 4; ++j) {
      v16bf kf = load_frag(Kk + ((size_t)bh * SEQ + t0 + 16) * DKDIM + j * 32, DKDIM);
      s1 = wmma_bf16(qf[j], kf, s1);
    }

    // --- online softmax; write P (bf16) to this wave's LDS tile ---
    const int col0 = t0 + r;
    const int col1 = t0 + 16 + r;
#pragma unroll
    for (int i = 0; i < 8; ++i) {
      const int rowg = q0 + i + h8;
      float a  = (col0 <= rowg) ? s0[i] * sc : -1e30f;
      float b2 = (col1 <= rowg) ? s1[i] * sc : -1e30f;
      float cmax  = rmax16(fmaxf(a, b2));
      float mnew  = fmaxf(m[i], cmax);
      float scale = __expf(m[i] - mnew);
      float p0 = __expf(a - mnew);
      float p1 = __expf(b2 - mnew);
      l[i] = l[i] * scale + rsum16(p0 + p1);
      m[i] = mnew;
#pragma unroll
      for (int j = 0; j < 8; ++j) of[j][i] *= scale;
      pbuf[wave][(i + h8) * 32 + r]      = f2bf(p0);
      pbuf[wave][(i + h8) * 32 + 16 + r] = f2bf(p1);
    }
    asm volatile("s_wait_dscnt 0" ::: "memory");  // LDS stores visible to this wave

    // --- O += P (16x32) * V (32x128) ---
    v16bf pf = load_frag(&pbuf[wave][0], 32);
#pragma unroll
    for (int j = 0; j < 8; ++j) {
      v16bf vf = load_frag(Vt + ((size_t)bh * DKDIM + j * 16) * SEQ + t0, SEQ);
      of[j] = wmma_bf16(pf, vf, of[j]);
    }
  }

  // --- epilogue: O /= l, store bf16 [B,S,H*D] ---
  float inv[8];
#pragma unroll
  for (int i = 0; i < 8; ++i) inv[i] = 1.0f / l[i];
  const int b = bh / NHEADS;
  const int h = bh % NHEADS;
#pragma unroll
  for (int j = 0; j < 8; ++j) {
#pragma unroll
    for (int i = 0; i < 8; ++i) {
      const int rowg = q0 + i + h8;
      const int d = j * 16 + r;
      O[((size_t)b * SEQ + rowg) * D_MODEL + h * DKDIM + d] = f2bf(of[j][i] * inv[i]);
    }
  }
}

// ---------- launch ----------

extern "C" void kernel_launch(void* const* d_in, const int* in_sizes, int n_in,
                              void* d_out, int out_size, void* d_ws, size_t ws_size,
                              hipStream_t stream) {
  const float* x  = (const float*)d_in[0];
  const float* Wq = (const float*)d_in[1];
  const float* bq = (const float*)d_in[2];
  const float* Wk = (const float*)d_in[3];
  const float* bk = (const float*)d_in[4];
  const float* Wv = (const float*)d_in[5];
  const float* bv = (const float*)d_in[6];
  const float* Wo = (const float*)d_in[7];
  const float* bo = (const float*)d_in[8];

  const size_t nx = (size_t)MROWS * D_MODEL;      // 8,388,608
  const size_t nw = (size_t)D_MODEL * D_MODEL;    // 4,194,304

  char* ws = (char*)d_ws;
  bf16* xb  = (bf16*)ws; ws += nx * sizeof(bf16);
  bf16* wqb = (bf16*)ws; ws += nw * sizeof(bf16);
  bf16* wkb = (bf16*)ws; ws += nw * sizeof(bf16);
  bf16* wvb = (bf16*)ws; ws += nw * sizeof(bf16);
  bf16* wob = (bf16*)ws; ws += nw * sizeof(bf16);
  bf16* Qb  = (bf16*)ws; ws += nx * sizeof(bf16);
  bf16* Kb  = (bf16*)ws; ws += nx * sizeof(bf16);
  bf16* Vt  = (bf16*)ws; ws += nx * sizeof(bf16);
  bf16* Ob  = (bf16*)ws; ws += nx * sizeof(bf16);

  // 1) convert inputs to bf16
  f32_to_bf16_kernel<<<4096, 256, 0, stream>>>(x,  xb,  (int)nx);
  f32_to_bf16_kernel<<<2048, 256, 0, stream>>>(Wq, wqb, (int)nw);
  f32_to_bf16_kernel<<<2048, 256, 0, stream>>>(Wk, wkb, (int)nw);
  f32_to_bf16_kernel<<<2048, 256, 0, stream>>>(Wv, wvb, (int)nw);
  f32_to_bf16_kernel<<<2048, 256, 0, stream>>>(Wo, wob, (int)nw);

  // 2) projections (Q,K into [B,H,S,D]; V transposed into [B,H,D,S])
  dim3 gg(D_MODEL / 256, MROWS / 128);            // (8, 32)
  gemm_bf16_kernel<<<gg, 256, 0, stream>>>(xb, wqb, bq, nullptr, Qb,
                                           MROWS, D_MODEL, D_MODEL, 1);
  gemm_bf16_kernel<<<gg, 256, 0, stream>>>(xb, wkb, bk, nullptr, Kb,
                                           MROWS, D_MODEL, D_MODEL, 1);
  gemm_bf16_kernel<<<gg, 256, 0, stream>>>(xb, wvb, bv, nullptr, Vt,
                                           MROWS, D_MODEL, D_MODEL, 2);

  // 3) causal flash attention -> Ob [B,S,D_MODEL] bf16
  attn_kernel<<<(BATCH * NHEADS * (SEQ / 16)) / 8, 256, 0, stream>>>(Qb, Kb, Vt, Ob);

  // 4) output projection -> fp32 d_out
  gemm_bf16_kernel<<<gg, 256, 0, stream>>>(Ob, wob, bo, (float*)d_out, nullptr,
                                           MROWS, D_MODEL, D_MODEL, 0);
}
